// TopKAttention_27779848471220
// MI455X (gfx1250) — compile-verified, matrix operand order checked by reading, not run on hardware
//
#include <hip/hip_runtime.h>
#include <math.h>

// ---------------- problem constants ----------------
#define S_TOK 4096
#define Dm    1024
#define Hh    16
#define HDm   64
#define KSEL  64
#define CAND_CAP 256

typedef __attribute__((ext_vector_type(16))) __bf16 v16bf;
typedef __attribute__((ext_vector_type(8)))  __bf16 v8bf;
typedef __attribute__((ext_vector_type(8)))  float  v8f;

// Build a 16-element bf16 A/B fragment from two contiguous 16B chunks.
// 16-bit operand layout (ISA 7.12.2): lanes 0-15 hold K=[k0..k0+7],[k0+16..k0+23],
// lanes 16-31 hold K=[k0+8..k0+15],[k0+24..k0+31]  ->  chunk offsets hi*8 based.
__device__ __forceinline__ v16bf load_frag(const __bf16* rowbase, int c0, int c1) {
    v8bf lo = *(const v8bf*)(rowbase + c0);
    v8bf hi = *(const v8bf*)(rowbase + c1);
    return __builtin_shufflevector(lo, hi, 0,1,2,3,4,5,6,7,8,9,10,11,12,13,14,15);
}

// float -> order-preserving unsigned key (larger float => larger uint)
__device__ __forceinline__ unsigned monokey(float f) {
    unsigned u = __float_as_uint(f);
    return (u & 0x80000000u) ? ~u : (u | 0x80000000u);
}

// ---------------- fp32 -> bf16 cast ----------------
__global__ __launch_bounds__(256) void cast_bf16_kernel(const float* __restrict__ src,
                                                        __bf16* __restrict__ dst, int n4) {
    int id = blockIdx.x * 256 + threadIdx.x;
    if (id < n4) {
        float4 f = ((const float4*)src)[id];
        dst[4*id+0] = (__bf16)f.x;
        dst[4*id+1] = (__bf16)f.y;
        dst[4*id+2] = (__bf16)f.z;
        dst[4*id+3] = (__bf16)f.w;
    }
}

// ---------------- generic WMMA GEMM: C[M=4096, N=1024] = A @ W^T + bias ----------------
// mode 0: store bf16 head-major [H,4096,64], value scaled by qscale   (q / k)
// mode 2: store f32  head-major [H,4096,64]                           (v)
// mode 3: store f32  flat [4096,1024]                                 (final out)
__global__ __launch_bounds__(256) void gemm_wmma_kernel(const __bf16* __restrict__ A,
                                                        const __bf16* __restrict__ W,
                                                        const float*  __restrict__ bias,
                                                        void* __restrict__ out,
                                                        int mode, float qscale) {
    const int lane = threadIdx.x & 31;
    const int wave = threadIdx.x >> 5;
    const int tile = blockIdx.x * 8 + wave;      // 16384 16x16 tiles total
    const int m0 = (tile >> 6) << 4;             // 64 N-tiles per M row
    const int n0 = (tile & 63) << 4;
    const int ln = lane & 15;
    const int hi = lane >> 4;

    const __bf16* arow = A + (size_t)(m0 + ln) * Dm;
    const __bf16* brow = W + (size_t)(n0 + ln) * Dm;

    v8f c = {};
    for (int k0 = 0; k0 < Dm; k0 += 32) {
        // locality=3 -> WGP-scope prefetch (pull into near cache; data is L2-resident)
        __builtin_prefetch(arow + k0 + 128, 0, 3);
        __builtin_prefetch(brow + k0 + 128, 0, 3);
        const int c0 = k0 + hi * 8;
        const int c1 = k0 + 16 + hi * 8;
        v16bf a = load_frag(arow, c0, c1);
        v16bf b = load_frag(brow, c0, c1);
        c = __builtin_amdgcn_wmma_f32_16x16x32_bf16(false, a, false, b, (short)0, c,
                                                    false, false);
    }

    // C layout: n = n0+ln fixed per lane; m = m0 + r + 8*hi
    const int n  = n0 + ln;
    const float bn = bias[n];

    if (mode == 3) {
        float* o = (float*)out;
        #pragma unroll
        for (int r = 0; r < 8; ++r)
            o[(size_t)(m0 + r + hi * 8) * Dm + n] = c[r] + bn;
    } else {
        const int h  = n >> 6;
        const int hd = n & 63;
        if (mode == 2) {
            float* vout = (float*)out;
            #pragma unroll
            for (int r = 0; r < 8; ++r) {
                int m = m0 + r + hi * 8;
                vout[((size_t)(h << 12) + m) * HDm + hd] = c[r] + bn;
            }
        } else {
            __bf16* qk = (__bf16*)out;
            #pragma unroll
            for (int r = 0; r < 8; ++r) {
                int m = m0 + r + hi * 8;
                qk[((size_t)(h << 12) + m) * HDm + hd] = (__bf16)((c[r] + bn) * qscale);
            }
        }
    }
}

// ---------------- scores + radix-select top-64 + softmax + gathered A*V ----------------
// One workgroup = one (head, 16-query tile). 256 threads = 8 waves.
// Dynamic LDS: 16x4096 f32 scores (256KB) + weights/indices + per-wave radix
// histograms and candidate buffers (~295KB total; CDNA5 WGP allows 320KB).
__global__ __launch_bounds__(256) void scores_topk_kernel(const __bf16* __restrict__ qbf,
                                                          const __bf16* __restrict__ kbf,
                                                          const float*  __restrict__ vf,
                                                          __bf16* __restrict__ attnbf,
                                                          int* __restrict__ idx_out) {
    extern __shared__ float smem[];
    float*          sc    = smem;                                   // [16][4096]
    volatile float* attnw = smem + 16 * S_TOK;                      // [16][64]
    volatile int*   idxb  = (volatile int*)(smem + 16 * S_TOK + 16 * KSEL);
    int*            hist  = (int*)(smem + 16 * S_TOK + 32 * KSEL);  // [8][256]
    float*          candv = (float*)(hist + 8 * 256);               // [8][CAND_CAP]
    int*            candi = (int*)(candv + 8 * CAND_CAP);           // [8][CAND_CAP]
    volatile int*   ccnt  = (volatile int*)(candi + 8 * CAND_CAP);  // [8]

    const int lane = threadIdx.x & 31;
    const int wave = threadIdx.x >> 5;
    const int h  = blockIdx.x >> 8;
    const int q0 = (blockIdx.x & 255) << 4;
    const int ln = lane & 15;
    const int hi = lane >> 4;

    const __bf16* qh = qbf + (size_t)(h << 12) * HDm;
    const __bf16* kh = kbf + (size_t)(h << 12) * HDm;

    // ---- Phase 1: 16x4096 score strip via WMMA (q pre-scaled by 1/sqrt(HD)) ----
    const __bf16* arow = qh + (size_t)(q0 + ln) * HDm;
    v16bf a0 = load_frag(arow,       hi * 8, 16 + hi * 8);
    v16bf a1 = load_frag(arow, 32 +  hi * 8, 48 + hi * 8);

    for (int t = wave; t < 256; t += 8) {
        const int kt0 = t << 4;
        const __bf16* brow = kh + (size_t)(kt0 + ln) * HDm;
        v16bf b0 = load_frag(brow,       hi * 8, 16 + hi * 8);
        v16bf b1 = load_frag(brow, 32 +  hi * 8, 48 + hi * 8);
        v8f c = {};
        c = __builtin_amdgcn_wmma_f32_16x16x32_bf16(false, a0, false, b0, (short)0, c,
                                                    false, false);
        c = __builtin_amdgcn_wmma_f32_16x16x32_bf16(false, a1, false, b1, (short)0, c,
                                                    false, false);
        #pragma unroll
        for (int r = 0; r < 8; ++r)
            sc[(r + hi * 8) * S_TOK + kt0 + ln] = c[r];
    }
    __syncthreads();

    int*          whist = hist  + wave * 256;
    float*        wcv   = candv + wave * CAND_CAP;
    int*          wci   = candi + wave * CAND_CAP;
    volatile int* wcnt  = ccnt  + wave;

    // ---- Phase 2+3: each wave owns queries 2w, 2w+1 (sc is read-only now) ----
    for (int sub = 0; sub < 2; ++sub) {
        const int ql = wave * 2 + sub;
        const float* row = sc + ql * S_TOK;

        // -- radix-select the 64th-largest key (exact threshold T + tie budget) --
        unsigned prefix = 0;
        int remaining = KSEL;
        for (int p = 0; p < 4; ++p) {
            for (int b = lane; b < 256; b += 32) whist[b] = 0;
            __asm__ volatile("" ::: "memory");
            const int shm = (32 - 8 * p) & 31;   // match shift (unused when p==0)
            const int shb = 24 - 8 * p;          // bin shift
            for (int j = lane; j < S_TOK; j += 32) {
                unsigned u = monokey(row[j]);
                if (p == 0 || (u >> shm) == prefix)
                    atomicAdd(&whist[(u >> shb) & 255], 1);
            }
            __asm__ volatile("" ::: "memory");
            // suffix scan over 256 bins: lane l owns bins [8l, 8l+7]
            int lc[8], s = 0;
            #pragma unroll
            for (int i = 0; i < 8; ++i) { lc[i] = whist[lane * 8 + i]; s += lc[i]; }
            int t = s;
            #pragma unroll
            for (int off = 1; off < 32; off <<= 1) {
                int ot = __shfl_down(t, off, 32);
                if (lane + off < 32) t += ot;
            }
            int cum = t - s;                     // total strictly above this lane's bins
            int packed = -1;
            #pragma unroll
            for (int i = 7; i >= 0; --i) {
                int cb = cum;
                cum += lc[i];
                if (cb < remaining && cum >= remaining && packed < 0)
                    packed = (cb << 8) | (lane * 8 + i);   // unique crossing bin
            }
            #pragma unroll
            for (int off = 16; off >= 1; off >>= 1)
                packed = max(packed, __shfl_xor(packed, off, 32));
            prefix = (prefix << 8) | (unsigned)(packed & 255);
            remaining -= (packed >> 8);
        }
        const unsigned T = prefix;   // exact monotonic key of the 64th largest

        // -- collect candidates >= T (all > T plus every tie == T) --
        if (lane == 0) *wcnt = 0;
        __asm__ volatile("" ::: "memory");
        for (int j = lane; j < S_TOK; j += 32) {
            float f = row[j];
            if (monokey(f) >= T) {
                int pos = atomicAdd((int*)wcnt, 1);
                if (pos < CAND_CAP) { wcv[pos] = f; wci[pos] = j; }
            }
        }
        __asm__ volatile("" ::: "memory");
        int C = *wcnt; if (C > CAND_CAP) C = CAND_CAP;

        // -- ordered emit: argmax-with-removal over <=256 candidates.
        //    (value desc, index asc on ties  ==  lax.top_k ordering) --
        for (int it = 0; it < KSEL; ++it) {
            float best = -3.0e38f; int bi = 0x7fffffff; int bp = -1;
            for (int j = lane; j < C; j += 32) {
                float vv = wcv[j]; int ii = wci[j];
                if (vv > best || (vv == best && ii < bi)) { best = vv; bi = ii; bp = j; }
            }
            #pragma unroll
            for (int off = 16; off >= 1; off >>= 1) {
                float ov = __shfl_xor(best, off, 32);
                int   oi = __shfl_xor(bi,   off, 32);
                int   op = __shfl_xor(bp,   off, 32);
                if (ov > best || (ov == best && oi < bi)) { best = ov; bi = oi; bp = op; }
            }
            if (lane == 0) {
                wcv[bp] = -3.0e38f;                       // remove winner
                attnw[ql * KSEL + it] = best;             // raw score (desc order)
                idxb [ql * KSEL + it] = bi;
                idx_out[(((size_t)(h << 12) + (q0 + ql)) << 6) + it] = bi;
            }
            __asm__ volatile("" ::: "memory");
        }

        // -- softmax over 64 picks; attnw[ql][0] is the max (stable) --
        float smax = attnw[ql * KSEL];
        float e0 = __expf(attnw[ql * KSEL + lane]      - smax);
        float e1 = __expf(attnw[ql * KSEL + lane + 32] - smax);
        float psum = e0 + e1;
        #pragma unroll
        for (int off = 16; off >= 1; off >>= 1)
            psum += __shfl_xor(psum, off, 32);
        float inv = 1.0f / psum;
        attnw[ql * KSEL + lane]      = e0 * inv;
        attnw[ql * KSEL + lane + 32] = e1 * inv;

        // -- gathered weighted sum of V rows; lane covers dims 2*lane, 2*lane+1 --
        float acc0 = 0.f, acc1 = 0.f;
        const int d0 = lane * 2, d1 = lane * 2 + 1;
        for (int kk = 0; kk < KSEL; ++kk) {
            float w = attnw[ql * KSEL + kk];
            int   rI = idxb [ql * KSEL + kk];
            const float* vrow = vf + ((size_t)(h << 12) + rI) * HDm;
            acc0 += w * vrow[d0];
            acc1 += w * vrow[d1];
        }
        __bf16* orow = attnbf + (size_t)(q0 + ql) * Dm + h * HDm;
        orow[d0] = (__bf16)acc0;
        orow[d1] = (__bf16)acc1;
    }
}

// ---------------- host launcher ----------------
extern "C" void kernel_launch(void* const* d_in, const int* in_sizes, int n_in,
                              void* d_out, int out_size, void* d_ws, size_t ws_size,
                              hipStream_t stream) {
    (void)in_sizes; (void)n_in; (void)out_size; (void)ws_size;
    const float* x  = (const float*)d_in[0];
    const float* Wq = (const float*)d_in[1];
    const float* bq = (const float*)d_in[2];
    const float* Wk = (const float*)d_in[3];
    const float* bk = (const float*)d_in[4];
    const float* Wv = (const float*)d_in[5];
    const float* bv = (const float*)d_in[6];
    const float* Wo = (const float*)d_in[7];
    const float* bo = (const float*)d_in[8];

    char* ws = (char*)d_ws;
    __bf16* xbf = (__bf16*)ws;  ws += (size_t)S_TOK * Dm * 2;
    __bf16* wqb = (__bf16*)ws;  ws += (size_t)Dm * Dm * 2;
    __bf16* wkb = (__bf16*)ws;  ws += (size_t)Dm * Dm * 2;
    __bf16* wvb = (__bf16*)ws;  ws += (size_t)Dm * Dm * 2;
    __bf16* wob = (__bf16*)ws;  ws += (size_t)Dm * Dm * 2;
    __bf16* qbf = (__bf16*)ws;  ws += (size_t)Hh * S_TOK * HDm * 2;
    __bf16* kbf = (__bf16*)ws;  ws += (size_t)Hh * S_TOK * HDm * 2;
    float*  vfp = (float*)ws;   ws += (size_t)Hh * S_TOK * HDm * 4;
    __bf16* abf = (__bf16*)ws;  ws += (size_t)S_TOK * Dm * 2;

    float* out_f  = (float*)d_out;
    int*   idxout = (int*)(out_f + (size_t)S_TOK * Dm);   // tuple tail: [H,S,K] int32

    const float qscale = 0.125f;  // 1/sqrt(HD)

    // 1) casts
    cast_bf16_kernel<<<(S_TOK * Dm / 4 + 255) / 256, 256, 0, stream>>>(x,  xbf, S_TOK * Dm / 4);
    cast_bf16_kernel<<<(Dm * Dm / 4 + 255) / 256,    256, 0, stream>>>(Wq, wqb, Dm * Dm / 4);
    cast_bf16_kernel<<<(Dm * Dm / 4 + 255) / 256,    256, 0, stream>>>(Wk, wkb, Dm * Dm / 4);
    cast_bf16_kernel<<<(Dm * Dm / 4 + 255) / 256,    256, 0, stream>>>(Wv, wvb, Dm * Dm / 4);
    cast_bf16_kernel<<<(Dm * Dm / 4 + 255) / 256,    256, 0, stream>>>(Wo, wob, Dm * Dm / 4);

    // 2) projections: 16384 tiles / 8 waves = 2048 blocks
    gemm_wmma_kernel<<<2048, 256, 0, stream>>>(xbf, wqb, bq, qbf, 0, qscale);
    gemm_wmma_kernel<<<2048, 256, 0, stream>>>(xbf, wkb, bk, kbf, 0, 1.0f);
    gemm_wmma_kernel<<<2048, 256, 0, stream>>>(xbf, wvb, bv, vfp, 2, 1.0f);

    // 3) scores + radix-select top-k + softmax + gathered AV
    const size_t smemB = ((size_t)16 * S_TOK + 32 * KSEL + 8 * 256
                          + 8 * CAND_CAP * 2 + 8) * 4;   // ~295KB < 320KB WGP LDS
    scores_topk_kernel<<<Hh * (S_TOK / 16), 256, smemB, stream>>>(qbf, kbf, vfp, abf, idxout);

    // 4) output projection
    gemm_wmma_kernel<<<2048, 256, 0, stream>>>(abf, wob, bo, out_f, 3, 1.0f);
}
